// Attention_21208548507651
// MI455X (gfx1250) — matrix operand order repaired and checked by reading, not compiled, not run on hardware
//
#include <hip/hip_runtime.h>

typedef __attribute__((ext_vector_type(16))) _Float16 v16h;
typedef __attribute__((ext_vector_type(8)))  _Float16 v8h;
typedef __attribute__((ext_vector_type(8)))  float    v8f;

#define DIM   512
#define NTOK  4096   // 4 groups * 1024 tokens
#define NKEYS 1024

// ---------------------------------------------------------------------------
// WMMA fragment helpers (layouts per CDNA5 ISA §7.12.2, wave32)
// A (16x32 f16): lane<16 row M=lane, elems 0..7=K0..7, 8..15=K16..23
//                lane>=16 row M=lane-16, elems 0..7=K8..15, 8..15=K24..31
// B (32x16 f16, stored transposed as X[n][k]): lane<16 col N=lane, elems 0..7=K0..7, 8..15=K16..23
//                lane>=16 col N=lane-16, elems 0..7=K8..15, 8..15=K24..31
// C/D (16x16 f32): VGPR r: lane<16 -> (M=r, N=lane); lane>=16 -> (M=r+8, N=lane-16)
// ---------------------------------------------------------------------------
__device__ __forceinline__ v16h load_a_frag(const _Float16* base, int row0,
                                            int stride, int k0) {
  int lane = threadIdx.x & 31;
  const _Float16* p = base + (size_t)(row0 + (lane & 15)) * stride + k0 + ((lane >> 4) << 3);
  v8h lo = *(const v8h*)(p);
  v8h hi = *(const v8h*)(p + 16);
  v16h r;
#pragma unroll
  for (int i = 0; i < 8; ++i) { r[i] = lo[i]; r[i + 8] = hi[i]; }
  return r;
}

__device__ __forceinline__ v16h load_b_frag(const _Float16* baseT, int n0,
                                            int stride, int k0) {
  int lane = threadIdx.x & 31;
  const _Float16* p = baseT + (size_t)(n0 + (lane & 15)) * stride + k0 + ((lane >> 4) << 3);
  v8h lo = *(const v8h*)(p);
  v8h hi = *(const v8h*)(p + 16);
  v16h r;
#pragma unroll
  for (int i = 0; i < 8; ++i) { r[i] = lo[i]; r[i + 8] = hi[i]; }
  return r;
}

__device__ __forceinline__ v8f wmma_f16(v16h a, v16h b, v8f c) {
  return __builtin_amdgcn_wmma_f32_16x16x32_f16(false, a, false, b,
                                                (short)0, c, false, false);
}

// Async DMA: 16 bytes per lane, global -> LDS, tracked by ASYNCcnt
// (CDNA5 ISA GLOBAL_LOAD_ASYNC_TO_LDS_B128)
__device__ __forceinline__ void async_load_b128(const _Float16* gptr, _Float16* lptr) {
  unsigned lds_off = (unsigned)(size_t)lptr;   // flat addr low 32 bits = LDS offset
  asm volatile("global_load_async_to_lds_b128 %0, %1, off"
               :: "v"(lds_off), "v"(gptr) : "memory");
}

__device__ __forceinline__ void wait_asynccnt0() {
  asm volatile("s_wait_asynccnt 0x0" ::: "memory");
}

// ---------------------------------------------------------------------------
// 1) LayerNorm -> f16, one wave per row of 512
// ---------------------------------------------------------------------------
__global__ void ln_f16_kernel(const float* __restrict__ q, const float* __restrict__ k,
                              const float* __restrict__ v, const float* __restrict__ g,
                              const float* __restrict__ b, _Float16* __restrict__ out) {
  int wave = threadIdx.x >> 5, lane = threadIdx.x & 31;
  int row = blockIdx.x * 8 + wave;                    // 0..4095
  int t = blockIdx.y;                                 // 0:q 1:k 2:v
  const float* x = (t == 0) ? q : (t == 1) ? k : v;
  const float* xr = x + (size_t)row * DIM;
  float vals[16], s = 0.f, ss = 0.f;
#pragma unroll
  for (int i = 0; i < 16; ++i) {
    float u = xr[i * 32 + lane];
    vals[i] = u; s += u; ss += u * u;
  }
#pragma unroll
  for (int m = 16; m >= 1; m >>= 1) {
    s += __shfl_xor(s, m, 32);
    ss += __shfl_xor(ss, m, 32);
  }
  float mu = s * (1.0f / DIM);
  float var = ss * (1.0f / DIM) - mu * mu;
  float rstd = rsqrtf(var + 1e-5f);
  _Float16* orow = out + (size_t)t * NTOK * DIM + (size_t)row * DIM;
#pragma unroll
  for (int i = 0; i < 16; ++i) {
    int c = i * 32 + lane;
    orow[c] = (_Float16)((vals[i] - mu) * rstd * g[c] + b[c]);
  }
}

// ---------------------------------------------------------------------------
// 2) W_in / W_out -> transposed f16 (Wt[n][k] = W[k][n]) for B-fragments
// ---------------------------------------------------------------------------
__global__ void wt_f16_kernel(const float* __restrict__ Win, const float* __restrict__ Wout,
                              _Float16* __restrict__ WtIn, _Float16* __restrict__ WtOut) {
  int idx = blockIdx.x * blockDim.x + threadIdx.x;    // 0..262143 : n*512+k
  int n = idx >> 9, kk = idx & 511;
  const float* W = blockIdx.y ? Wout : Win;
  _Float16* T = blockIdx.y ? WtOut : WtIn;
  T[idx] = (_Float16)W[(size_t)kk * DIM + n];
}

// ---------------------------------------------------------------------------
// 3) Projection GEMM: [4096x512] x [512x512] -> f16, 4-wave blocks, 64x64 tile
// ---------------------------------------------------------------------------
__global__ __launch_bounds__(128) void proj_gemm_kernel(const _Float16* __restrict__ xln,
                                                        const _Float16* __restrict__ WtIn,
                                                        _Float16* __restrict__ out) {
  int w = threadIdx.x >> 5, lane = threadIdx.x & 31, lane15 = lane & 15;
  int zt = blockIdx.z;
  const _Float16* A = xln + (size_t)zt * NTOK * DIM;
  _Float16* O = out + (size_t)zt * NTOK * DIM;
  int m0 = blockIdx.x * 64 + w * 16;
  int n0 = blockIdx.y * 64;
  v8f acc[4] = {};
  for (int k0 = 0; k0 < DIM; k0 += 32) {
    if (k0 + 64 < DIM)
      __builtin_prefetch(A + (size_t)(m0 + lane15) * DIM + k0 + 64, 0, 0);
    v16h a = load_a_frag(A, m0, DIM, k0);
#pragma unroll
    for (int nt = 0; nt < 4; ++nt)
      acc[nt] = wmma_f16(a, load_b_frag(WtIn, n0 + nt * 16, DIM, k0), acc[nt]);
  }
#pragma unroll
  for (int nt = 0; nt < 4; ++nt)
#pragma unroll
    for (int r = 0; r < 8; ++r) {
      int row = m0 + ((lane < 16) ? r : r + 8);
      int col = n0 + nt * 16 + lane15;
      O[(size_t)row * DIM + col] = (_Float16)acc[nt][r];
    }
}

// ---------------------------------------------------------------------------
// 4) Repack V to head-dim-major: Vt[hg][d][n]
// ---------------------------------------------------------------------------
__global__ void vt_repack_kernel(const _Float16* __restrict__ Vh, _Float16* __restrict__ Vt) {
  int t = blockIdx.x * blockDim.x + threadIdx.x;      // 0..2M-1
  int hg = t >> 16, rem = t & 65535;
  int d = rem >> 10, n = rem & 1023;
  int h = hg >> 2, g = hg & 3;
  Vt[t] = Vh[(size_t)(g * 1024 + n) * DIM + h * 64 + d];
}

// ---------------------------------------------------------------------------
// 5) Per-token stats: ||fq||, mean_d(fq), ||fk||, sum_d(fk)  (index hg*1024+n)
// ---------------------------------------------------------------------------
__global__ void stats_kernel(const _Float16* __restrict__ Qh, const _Float16* __restrict__ Kh,
                             float* __restrict__ nq, float* __restrict__ muq,
                             float* __restrict__ nk, float* __restrict__ sk) {
  int t = blockIdx.x * blockDim.x + threadIdx.x;      // 0..32767
  int hg = t >> 10, n = t & 1023;
  int h = hg >> 2, g = hg & 3;
  const _Float16* qr = Qh + (size_t)(g * 1024 + n) * DIM + h * 64;
  const _Float16* kr = Kh + (size_t)(g * 1024 + n) * DIM + h * 64;
  float s = 0.f, ss = 0.f, s2 = 0.f, ss2 = 0.f;
  for (int d = 0; d < 64; ++d) {
    float a = (float)qr[d]; s += a;  ss += a * a;
    float b = (float)kr[d]; s2 += b; ss2 += b * b;
  }
  nq[t] = sqrtf(ss);  muq[t] = s * (1.0f / 64.0f);
  nk[t] = sqrtf(ss2); sk[t] = s2;
}

// ---------------------------------------------------------------------------
// 6) Flash attention, 64-row q-strip per block (4 waves x 16 rows), streaming
//    16 key tiles of 64 with async-DMA double-buffered K/V tiles in LDS.
//    var_comp and row-constant cov terms cancel under softmax:
//    score = 0.5*cosw*clip(dot/(nq*nk)) + 0.5*covw*0.01/8192*(dot - mu_q*s_k)
// ---------------------------------------------------------------------------
__global__ __launch_bounds__(128) void attn_kernel(
    const _Float16* __restrict__ Qh, const _Float16* __restrict__ Kh,
    const _Float16* __restrict__ Vt, _Float16* __restrict__ Oh,
    const float* __restrict__ nq_s, const float* __restrict__ mu_s,
    const float* __restrict__ nk_s, const float* __restrict__ sk_s,
    const float* __restrict__ cov_p, const float* __restrict__ var_p) {
  __shared__ alignas(32) _Float16 kbuf[2][64][64];   // K tile, row-major keys x d
  __shared__ alignas(32) _Float16 vbuf[2][64][64];   // V tile, d x keys
  __shared__ alignas(32) _Float16 p_lds[4][16][64];  // per-wave P strip
  int tid = threadIdx.x;
  int w = tid >> 5, lane = tid & 31, lane15 = lane & 15;
  int qtile = blockIdx.x;          // 0..15
  int hg = blockIdx.y;             // 0..31
  int h = hg >> 2, g = hg & 3;

  float covw = 1.0f / (1.0f + __expf(-cov_p[0]));
  float varw = 1.0f / (1.0f + __expf(-var_p[0]));
  float cosw = fminf(fmaxf(1.0f - covw - varw, 0.1f), 0.8f);
  const float COSW = cosw * 0.5f;
  const float COVC = covw * 0.1f * (0.1f / 1024.0f) * (1.0f / 8.0f) * 0.5f;

  const _Float16* Qb = Qh + (size_t)g * 1024 * DIM + h * 64;
  const _Float16* Kb = Kh + (size_t)g * 1024 * DIM + h * 64;
  const _Float16* Vb = Vt + (size_t)hg * 64 * 1024;

  int q0 = qtile * 64 + w * 16;
  int statQ = hg * 1024 + q0;

  // cooperative async stage of one 64x64 K tile + one 64x64 V tile
  auto stage_tile = [&](int kt, int buf) {
    int kb = kt * 64;
#pragma unroll
    for (int j = 0; j < 4; ++j) {
      int c = j * 128 + tid;          // 0..511 : 16-byte chunk id
      int row = c >> 3, col = (c & 7) * 8;
      async_load_b128(Kb + (size_t)(kb + row) * DIM + col, &kbuf[buf][row][col]);
      async_load_b128(Vb + (size_t)row * NKEYS + kb + col, &vbuf[buf][row][col]);
    }
  };

  v16h aq[2];
  aq[0] = load_a_frag(Qb, q0, DIM, 0);
  aq[1] = load_a_frag(Qb, q0, DIM, 32);

  float rnq[8], rmu[8], mrow[8], lrow[8];
#pragma unroll
  for (int r = 0; r < 8; ++r) {
    int rr = statQ + ((lane < 16) ? r : r + 8);
    rnq[r] = nq_s[rr] + 1e-8f;
    rmu[r] = mu_s[rr];
    mrow[r] = -1e30f;
    lrow[r] = 0.0f;
  }
  v8f o[4] = {};

  stage_tile(0, 0);
  wait_asynccnt0();
  __syncthreads();

  for (int kt = 0; kt < 16; ++kt) {
    int buf = kt & 1;
    if (kt + 1 < 16) stage_tile(kt + 1, buf ^ 1);   // prefetch next tile

    // ---- S = Q K^T (16x64 per wave) from LDS K tile ----
    v8f s[4];
#pragma unroll
    for (int nt = 0; nt < 4; ++nt) {
      v8f acc = {};
      acc = wmma_f16(aq[0], load_b_frag(&kbuf[buf][0][0], nt * 16, 64, 0), acc);
      acc = wmma_f16(aq[1], load_b_frag(&kbuf[buf][0][0], nt * 16, 64, 32), acc);
      s[nt] = acc;
    }
    // ---- score transform ----
    int kbase = kt * 64;
#pragma unroll
    for (int nt = 0; nt < 4; ++nt) {
      int m = hg * 1024 + kbase + nt * 16 + lane15;
      float nkv = nk_s[m] + 1e-8f;
      float skv = sk_s[m];
#pragma unroll
      for (int r = 0; r < 8; ++r) {
        float d = s[nt][r];
        float cs = d / (rnq[r] * nkv);
        cs = fminf(fmaxf(cs, -1.0f + 1e-8f), 1.0f - 1e-8f);
        s[nt][r] = COSW * cs + COVC * (d - rmu[r] * skv);
      }
    }
    // ---- online softmax: row max over 64 cols ----
    float tmax[8];
#pragma unroll
    for (int r = 0; r < 8; ++r) {
      float m4 = fmaxf(fmaxf(s[0][r], s[1][r]), fmaxf(s[2][r], s[3][r]));
#pragma unroll
      for (int msk = 8; msk >= 1; msk >>= 1) m4 = fmaxf(m4, __shfl_xor(m4, msk, 32));
      tmax[r] = m4;
    }
    float lsum[8];
#pragma unroll
    for (int r = 0; r < 8; ++r) {
      float mnew = fmaxf(mrow[r], tmax[r]);
      float scale = __expf(mrow[r] - mnew);
      mrow[r] = mnew;
      lrow[r] *= scale;
      lsum[r] = 0.0f;
#pragma unroll
      for (int t = 0; t < 4; ++t) o[t][r] *= scale;
    }
    int prow = (lane < 16) ? 0 : 8;
#pragma unroll
    for (int nt = 0; nt < 4; ++nt) {
#pragma unroll
      for (int r = 0; r < 8; ++r) {
        float p = __expf(s[nt][r] - mrow[r]);
        lsum[r] += p;
        p_lds[w][prow + r][nt * 16 + lane15] = (_Float16)p;
      }
    }
#pragma unroll
    for (int r = 0; r < 8; ++r) {
      float l = lsum[r];
#pragma unroll
      for (int msk = 8; msk >= 1; msk >>= 1) l += __shfl_xor(l, msk, 32);
      lrow[r] += l;
    }
    // per-wave LDS P strip: enforce store->load ordering (same-wave, in-order DS)
    asm volatile("s_wait_dscnt 0" ::: "memory");
    // ---- O += P @ V from LDS V tile ----
#pragma unroll
    for (int kk = 0; kk < 2; ++kk) {
      v16h ap = load_a_frag(&p_lds[w][0][0], 0, 64, kk * 32);
#pragma unroll
      for (int t = 0; t < 4; ++t)
        o[t] = wmma_f16(ap, load_b_frag(&vbuf[buf][0][0], t * 16, 64, kk * 32), o[t]);
    }
    // next tile's async loads must land (and this tile's LDS reads retire)
    // before anyone overwrites buffers
    wait_asynccnt0();
    __syncthreads();
  }
  // ---- normalize + store f16, layout [g*1024+n][h*64+d] ----
#pragma unroll
  for (int r = 0; r < 8; ++r) {
    int rowLocal = q0 + ((lane < 16) ? r : r + 8);
    float inv = 1.0f / lrow[r];
    size_t rowOff = (size_t)(g * 1024 + rowLocal) * DIM + h * 64;
#pragma unroll
    for (int t = 0; t < 4; ++t)
      Oh[rowOff + t * 16 + lane15] = (_Float16)(o[t][r] * inv);
  }
}

// ---------------------------------------------------------------------------
// 7) Output GEMM: [4096x512]f16 x [512x512]f16 + bias -> f32 d_out
// ---------------------------------------------------------------------------
__global__ __launch_bounds__(128) void out_gemm_kernel(const _Float16* __restrict__ Oh,
                                                       const _Float16* __restrict__ WtOut,
                                                       const float* __restrict__ b_out,
                                                       float* __restrict__ out) {
  int w = threadIdx.x >> 5, lane = threadIdx.x & 31, lane15 = lane & 15;
  int m0 = blockIdx.x * 64 + w * 16;
  int n0 = blockIdx.y * 64;
  v8f acc[4] = {};
  for (int k0 = 0; k0 < DIM; k0 += 32) {
    if (k0 + 64 < DIM)
      __builtin_prefetch(Oh + (size_t)(m0 + lane15) * DIM + k0 + 64, 0, 0);
    v16h a = load_a_frag(Oh, m0, DIM, k0);
#pragma unroll
    for (int nt = 0; nt < 4; ++nt)
      acc[nt] = wmma_f16(a, load_b_frag(WtOut, n0 + nt * 16, DIM, k0), acc[nt]);
  }
#pragma unroll
  for (int nt = 0; nt < 4; ++nt) {
    int col = n0 + nt * 16 + lane15;
    float bias = b_out[col];
#pragma unroll
    for (int r = 0; r < 8; ++r) {
      int row = m0 + ((lane < 16) ? r : r + 8);
      out[(size_t)row * DIM + col] = acc[nt][r] + bias;
    }
  }
}

// ---------------------------------------------------------------------------
extern "C" void kernel_launch(void* const* d_in, const int* in_sizes, int n_in,
                              void* d_out, int out_size, void* d_ws, size_t ws_size,
                              hipStream_t stream) {
  (void)in_sizes; (void)n_in; (void)out_size; (void)ws_size;
  const float* q     = (const float*)d_in[0];
  const float* k     = (const float*)d_in[1];
  const float* v     = (const float*)d_in[2];
  const float* ln_g  = (const float*)d_in[3];
  const float* ln_b  = (const float*)d_in[4];
  const float* W_in  = (const float*)d_in[5];
  const float* cov_p = (const float*)d_in[6];
  const float* var_p = (const float*)d_in[7];
  const float* W_out = (const float*)d_in[8];
  const float* b_out = (const float*)d_in[9];

  char* ws = (char*)d_ws;
  size_t off = 0;
  auto wsalloc = [&](size_t bytes) -> char* {
    char* p = ws + off;
    off += (bytes + 255) & ~(size_t)255;
    return p;
  };
  _Float16* xln   = (_Float16*)wsalloc(3ull * NTOK * DIM * 2);   // LN outputs (q,k,v)
  _Float16* QKVh  = (_Float16*)wsalloc(3ull * NTOK * DIM * 2);   // projected f16 (contiguous)
  _Float16* WtIn  = (_Float16*)wsalloc((size_t)DIM * DIM * 2);
  _Float16* WtOut = (_Float16*)wsalloc((size_t)DIM * DIM * 2);
  _Float16* Vt    = (_Float16*)wsalloc(32ull * 64 * 1024 * 2);   // V head-dim-major
  _Float16* Oh    = (_Float16*)wsalloc((size_t)NTOK * DIM * 2);  // attention output f16
  float* nq = (float*)wsalloc(32768ull * 4);
  float* mu = (float*)wsalloc(32768ull * 4);
  float* nk = (float*)wsalloc(32768ull * 4);
  float* sk = (float*)wsalloc(32768ull * 4);

  _Float16* Qh = QKVh;
  _Float16* Kh = QKVh + (size_t)NTOK * DIM;
  _Float16* Vh = QKVh + 2ull * NTOK * DIM;

  wt_f16_kernel<<<dim3(1024, 2), 256, 0, stream>>>(W_in, W_out, WtIn, WtOut);
  ln_f16_kernel<<<dim3(512, 3), 256, 0, stream>>>(q, k, v, ln_g, ln_b, xln);
  proj_gemm_kernel<<<dim3(64, 8, 3), 128, 0, stream>>>(xln, WtIn, QKVh);
  vt_repack_kernel<<<8192, 256, 0, stream>>>(Vh, Vt);
  stats_kernel<<<128, 256, 0, stream>>>(Qh, Kh, nq, mu, nk, sk);
  attn_kernel<<<dim3(16, 32), 128, 0, stream>>>(Qh, Kh, Vt, Oh, nq, mu, nk, sk, cov_p, var_p);
  out_gemm_kernel<<<dim3(64, 8), 128, 0, stream>>>(Oh, WtOut, b_out, (float*)d_out);
}
